// Attention_64003602645548
// MI455X (gfx1250) — matrix operand order repaired and checked by reading, not compiled, not run on hardware
//
#include <hip/hip_runtime.h>
#include <hip/hip_bf16.h>
#include <math.h>

// Problem dims (fixed by the reference).
#define Bb 32
#define Ss 4096
#define Dd 512
#define Hh 512
// Padded LDS row stride (bf16 elems): 520*2B = 1040B = 260 dwords -> 4-bank rotation per row,
// 16B-aligned rows, conflict-free 16-lane ds_load_b128 fragment reads.
#define SD2 520

typedef __bf16 bf16_t;
typedef bf16_t v16bf  __attribute__((ext_vector_type(16)));
typedef bf16_t bf16x8 __attribute__((ext_vector_type(8)));
typedef float  v8f    __attribute__((ext_vector_type(8)));

// ---------------------------------------------------------------- K0: W_ctx -> bf16 hi/lo split
__global__ __launch_bounds__(256) void cvt_wctx_kernel(const float* __restrict__ w,
                                                       bf16_t* __restrict__ hi,
                                                       bf16_t* __restrict__ lo, int n) {
  int i = blockIdx.x * 256 + threadIdx.x;
  if (i < n) {
    float x = w[i];
    bf16_t h = (bf16_t)x;
    hi[i] = h;
    lo[i] = (bf16_t)(x - (float)h);
  }
}

// ---------------------------------------------------------------- K1: g[b,h] = input@W_in.T + b_in + b_ctx
__global__ __launch_bounds__(128) void g_kernel(const float* __restrict__ input,
                                                const float* __restrict__ Win,
                                                const float* __restrict__ bin,
                                                const float* __restrict__ bctx,
                                                float* __restrict__ g) {
  int b = blockIdx.x;
  int h = blockIdx.y * 128 + threadIdx.x;
  const float* x = input + b * Dd;
  const float* w = Win + (size_t)h * Dd;
  float acc = bin[h] + bctx[h];
#pragma unroll 4
  for (int d = 0; d < Dd; ++d) acc += x[d] * w[d];
  g[b * Hh + h] = acc;
}

// ---------------------------------------------------------------- K2: fused WMMA GEMM + tanh + V-dot
// 256-thread block (8 waves) per (b, 32-row s-tile). Context tile staged in dynamic LDS as bf16
// hi/lo planes (2 x 16-row halves). Each wave owns 4 h-tiles; per K-chunk one B hi/lo fragment
// pair is reused across BOTH 16-row A tiles -> 6 WMMAs per B load (2x arithmetic intensity on
// the W_ctx stream vs. single-M tiling).
__global__ __launch_bounds__(256) void ctx_att_kernel(const float* __restrict__ ctxg,
                                                      const bf16_t* __restrict__ wctx_hi,
                                                      const bf16_t* __restrict__ wctx_lo,
                                                      const float* __restrict__ g,
                                                      const float* __restrict__ V,
                                                      float* __restrict__ att) {
  extern __shared__ char smem_raw[];
  bf16_t* sAh  = (bf16_t*)smem_raw;          // 32 x SD2 bf16  (hi plane)
  bf16_t* sAl  = sAh + 32 * SD2;             // 32 x SD2 bf16  (lo plane)
  float* att_w = (float*)(sAl + 32 * SD2);   // [8 waves][32 s-rows]

  int st = blockIdx.x, b = blockIdx.y;
  int s0 = st * 32;
  int tid = threadIdx.x;

  // Cooperative staging: 32 x 512 fp32 context rows -> bf16 hi/lo planes in LDS.
  const float* src = ctxg + ((size_t)b * Ss + s0) * Dd;
  for (int idx = tid; idx < 32 * (Dd / 4); idx += 256) {
    int r = idx >> 7;              // Dd/4 == 128
    int c = (idx & 127) * 4;
    float4 v = *(const float4*)(src + r * Dd + c);
    bf16_t* dh = sAh + r * SD2 + c;
    bf16_t* dl = sAl + r * SD2 + c;
    bf16_t h0 = (bf16_t)v.x, h1 = (bf16_t)v.y, h2 = (bf16_t)v.z, h3 = (bf16_t)v.w;
    dh[0] = h0; dh[1] = h1; dh[2] = h2; dh[3] = h3;
    dl[0] = (bf16_t)(v.x - (float)h0);
    dl[1] = (bf16_t)(v.y - (float)h1);
    dl[2] = (bf16_t)(v.z - (float)h2);
    dl[3] = (bf16_t)(v.w - (float)h3);
  }
  __syncthreads();

  int lane = tid & 31, wv = tid >> 5;
  int nlo = lane & 15, hhalf = lane >> 4;

  float attp0[8], attp1[8];
#pragma unroll
  for (int r = 0; r < 8; ++r) { attp0[r] = 0.f; attp1[r] = 0.f; }

  // A fragment bases (contiguous K order, matching B): lanes 0-15 take K {0..15},
  // lanes 16-31 take K {16..31}. Tile1 = rows 16..31 of the s-tile.
  const bf16_t* a0h = sAh + nlo * SD2 + hhalf * 16;
  const bf16_t* a0l = sAl + nlo * SD2 + hhalf * 16;
  const bf16_t* a1h = a0h + 16 * SD2;
  const bf16_t* a1l = a0l + 16 * SD2;

#pragma unroll 1
  for (int i = 0; i < 4; ++i) {
    int h0 = (wv + i * 8) << 4;
    // B fragment (32x16): lane holds column h0+nlo over 16 contiguous K values.
    const bf16_t* browh = wctx_hi + (size_t)(h0 + nlo) * Dd + hhalf * 16;
    const bf16_t* browl = wctx_lo + (size_t)(h0 + nlo) * Dd + hhalf * 16;

    v8f acc0 = {0.f, 0.f, 0.f, 0.f, 0.f, 0.f, 0.f, 0.f};
    v8f acc1 = {0.f, 0.f, 0.f, 0.f, 0.f, 0.f, 0.f, 0.f};
#pragma unroll 2
    for (int dc = 0; dc < Dd; dc += 32) {
      v16bf bh = *(const v16bf*)(browh + dc);
      v16bf bl = *(const v16bf*)(browl + dc);

      bf16x8 t0 = *(const bf16x8*)(a0h + dc);
      bf16x8 t1 = *(const bf16x8*)(a0h + dc + 8);
      v16bf A0h = __builtin_shufflevector(t0, t1, 0,1,2,3,4,5,6,7,8,9,10,11,12,13,14,15);
      t0 = *(const bf16x8*)(a0l + dc);
      t1 = *(const bf16x8*)(a0l + dc + 8);
      v16bf A0l = __builtin_shufflevector(t0, t1, 0,1,2,3,4,5,6,7,8,9,10,11,12,13,14,15);
      t0 = *(const bf16x8*)(a1h + dc);
      t1 = *(const bf16x8*)(a1h + dc + 8);
      v16bf A1h = __builtin_shufflevector(t0, t1, 0,1,2,3,4,5,6,7,8,9,10,11,12,13,14,15);
      t0 = *(const bf16x8*)(a1l + dc);
      t1 = *(const bf16x8*)(a1l + dc + 8);
      v16bf A1l = __builtin_shufflevector(t0, t1, 0,1,2,3,4,5,6,7,8,9,10,11,12,13,14,15);

      acc0 = __builtin_amdgcn_wmma_f32_16x16x32_bf16(false, A0l, false, bh, (short)0, acc0, false, false);
      acc0 = __builtin_amdgcn_wmma_f32_16x16x32_bf16(false, A0h, false, bl, (short)0, acc0, false, false);
      acc0 = __builtin_amdgcn_wmma_f32_16x16x32_bf16(false, A0h, false, bh, (short)0, acc0, false, false);
      acc1 = __builtin_amdgcn_wmma_f32_16x16x32_bf16(false, A1l, false, bh, (short)0, acc1, false, false);
      acc1 = __builtin_amdgcn_wmma_f32_16x16x32_bf16(false, A1h, false, bl, (short)0, acc1, false, false);
      acc1 = __builtin_amdgcn_wmma_f32_16x16x32_bf16(false, A1h, false, bh, (short)0, acc1, false, false);
    }
    // C layout: acc[r] = ctx[s0 + r + 8*hhalf (+16 for acc1)][h0 + nlo]; fold tanh + V dot over h.
    float vh = V[h0 + nlo];
    float gh = g[b * Hh + h0 + nlo];
#pragma unroll
    for (int r = 0; r < 8; ++r) {
      attp0[r] += vh * tanhf(gh + acc0[r]);
      attp1[r] += vh * tanhf(gh + acc1[r]);
    }
  }

  // Reduce across the 16 lanes of each half (same s rows), then deterministic cross-wave sum.
#pragma unroll
  for (int r = 0; r < 8; ++r) {
    float v0 = attp0[r], v1 = attp1[r];
    v0 += __shfl_xor(v0, 1, 32); v1 += __shfl_xor(v1, 1, 32);
    v0 += __shfl_xor(v0, 2, 32); v1 += __shfl_xor(v1, 2, 32);
    v0 += __shfl_xor(v0, 4, 32); v1 += __shfl_xor(v1, 4, 32);
    v0 += __shfl_xor(v0, 8, 32); v1 += __shfl_xor(v1, 8, 32);
    if (nlo == 0) {
      att_w[wv * 32 + hhalf * 8 + r]      = v0;
      att_w[wv * 32 + 16 + hhalf * 8 + r] = v1;
    }
  }
  __syncthreads();
  if (tid < 32) {
    float s = 0.f;
#pragma unroll
    for (int w = 0; w < 8; ++w) s += att_w[w * 32 + tid];
    att[(size_t)b * Ss + s0 + tid] = s;
  }
}

// ---------------------------------------------------------------- K3: masked softmax over S, one block per b
__global__ __launch_bounds__(256) void softmax_kernel(const float* __restrict__ att,
                                                      const unsigned char* __restrict__ mask,
                                                      float* __restrict__ alpha,
                                                      float* __restrict__ alpha_out) {
  __shared__ float red[256];
  int b = blockIdx.x, tid = threadIdx.x;
  const float* ab = att + (size_t)b * Ss;
  const unsigned char* mb = mask + (size_t)b * Ss;

  float m = -INFINITY;
  for (int s = tid; s < Ss; s += 256)
    if (!mb[s]) m = fmaxf(m, ab[s]);
  red[tid] = m; __syncthreads();
  for (int o = 128; o > 0; o >>= 1) { if (tid < o) red[tid] = fmaxf(red[tid], red[tid + o]); __syncthreads(); }
  m = red[0]; __syncthreads();

  float* alb = alpha + (size_t)b * Ss;
  float sum = 0.f;
  for (int s = tid; s < Ss; s += 256) {
    float e = mb[s] ? 0.f : expf(ab[s] - m);
    alb[s] = e;
    sum += e;
  }
  red[tid] = sum; __syncthreads();
  for (int o = 128; o > 0; o >>= 1) { if (tid < o) red[tid] += red[tid + o]; __syncthreads(); }
  float inv = 1.f / red[0];

  float* ob = alpha_out + (size_t)b * Ss;
  for (int s = tid; s < Ss; s += 256) {
    float a = alb[s] * inv;
    alb[s] = a;
    ob[s] = a + 1e-6f;
  }
}

// ---------------------------------------------------------------- K4: cbar partials = sum_s alpha*context
// Memory-bound 256 MB stream: float4 per lane, 128 threads cover a full 2KB context row per
// iteration; S split 32 ways -> 1024 blocks of outstanding b128 loads to fill 23.3 TB/s.
__global__ __launch_bounds__(128) void cbar_part_kernel(const float* __restrict__ ctxg,
                                                        const float* __restrict__ alpha,
                                                        float* __restrict__ part) {
  int b = blockIdx.x, sc = blockIdx.y;
  int tid = threadIdx.x;
  int sbeg = sc * 128;
  const float* base = ctxg + ((size_t)b * Ss + sbeg) * Dd + tid * 4;
  const float* al = alpha + (size_t)b * Ss + sbeg;
  float4 acc = {0.f, 0.f, 0.f, 0.f};
#pragma unroll 4
  for (int s = 0; s < 128; ++s) {
    float a = al[s];
    float4 v = *(const float4*)(base + (size_t)s * Dd);
    acc.x += a * v.x; acc.y += a * v.y; acc.z += a * v.z; acc.w += a * v.w;
  }
  float* dst = part + ((size_t)sc * Bb + b) * Dd + tid * 4;
  dst[0] = acc.x; dst[1] = acc.y; dst[2] = acc.z; dst[3] = acc.w;
}

__global__ __launch_bounds__(256) void cbar_reduce_kernel(const float* __restrict__ part,
                                                          float* __restrict__ cbar) {
  int i = blockIdx.x * 256 + threadIdx.x;
  float a = 0.f;
#pragma unroll
  for (int p = 0; p < 32; ++p) a += part[(size_t)p * Bb * Dd + i];
  cbar[i] = a;
}

// ---------------------------------------------------------------- K5: hidden = cbar @ W_ctx.T + b_ctx  (uses sum alpha == 1)
__global__ __launch_bounds__(128) void hidden_kernel(const float* __restrict__ cbar,
                                                     const float* __restrict__ Wctx,
                                                     const float* __restrict__ bctx,
                                                     float* __restrict__ out) {
  int b = blockIdx.x;
  int h = blockIdx.y * 128 + threadIdx.x;
  const float* c = cbar + b * Dd;
  const float* w = Wctx + (size_t)h * Dd;
  float acc = bctx[h];
#pragma unroll 4
  for (int d = 0; d < Dd; ++d) acc += c[d] * w[d];
  out[b * Hh + h] = acc;
}

// ----------------------------------------------------------------
extern "C" void kernel_launch(void* const* d_in, const int* in_sizes, int n_in,
                              void* d_out, int out_size, void* d_ws, size_t ws_size,
                              hipStream_t stream) {
  const float* input   = (const float*)d_in[0];
  const float* context = (const float*)d_in[1];
  const unsigned char* mask = (const unsigned char*)d_in[2];  // jax bool -> 1 byte/elem
  const float* W_in  = (const float*)d_in[3];
  const float* b_in  = (const float*)d_in[4];
  const float* W_ctx = (const float*)d_in[5];
  const float* b_ctx = (const float*)d_in[6];
  const float* V     = (const float*)d_in[7];

  float* out_hidden = (float*)d_out;                 // [B,H] first
  float* out_alpha  = out_hidden + Bb * Hh;          // then [B,S]

  char* ws = (char*)d_ws;
  bf16_t* wctx_hi = (bf16_t*)(ws + 0);               //  512 KB
  bf16_t* wctx_lo = (bf16_t*)(ws + 524288);          //  512 KB
  float*  g       = (float*)(ws + 1048576);          //   64 KB
  float*  att     = (float*)(ws + 1114112);          //  512 KB
  float*  alpha   = (float*)(ws + 1638400);          //  512 KB
  float*  part    = (float*)(ws + 2162688);          //    2 MB (32 partials x B x D)
  float*  cbar    = (float*)(ws + 4259840);          //   64 KB   (total ~4.3 MB)

  // Dynamic LDS for K2: 2 planes x 32 rows x SD2 bf16 + 8x32 float partials = 67,584 B
  // (> 64KB static limit, fine as dynamic: WGP owns 320KB -> 4 concurrent WGs).
  const size_t k2_smem = (size_t)2 * 32 * SD2 * sizeof(bf16_t) + 8 * 32 * sizeof(float);

  cvt_wctx_kernel<<<(Hh * Dd + 255) / 256, 256, 0, stream>>>(W_ctx, wctx_hi, wctx_lo, Hh * Dd);
  g_kernel<<<dim3(Bb, Hh / 128), 128, 0, stream>>>(input, W_in, b_in, b_ctx, g);
  ctx_att_kernel<<<dim3(Ss / 32, Bb), 256, k2_smem, stream>>>(context, wctx_hi, wctx_lo, g, V, att);
  softmax_kernel<<<Bb, 256, 0, stream>>>(att, mask, alpha, out_alpha);
  cbar_part_kernel<<<dim3(Bb, 32), 128, 0, stream>>>(context, alpha, part);
  cbar_reduce_kernel<<<(Bb * Dd) / 256, 256, 0, stream>>>(part, cbar);
  hidden_kernel<<<dim3(Bb, Hh / 128), 128, 0, stream>>>(cbar, W_ctx, b_ctx, out_hidden);
}
